// GAE_48378511622553
// MI455X (gfx1250) — matrix operand order repaired and checked by reading, not compiled, not run on hardware
//
#include <hip/hip_runtime.h>
#include <stdint.h>

#define N_NODES 50000
#define N_EDGES 800000
#define H       128
#define NLAYERS 2
#define NOUT    3

typedef _Float16 v16h  __attribute__((ext_vector_type(16)));
typedef float    v8f   __attribute__((ext_vector_type(8)));
typedef uint32_t u32x4 __attribute__((ext_vector_type(4)));
typedef float    f4    __attribute__((ext_vector_type(4)));
typedef _Float16 h4    __attribute__((ext_vector_type(4)));

union FragH { v16h v; u32x4 q[2]; };

__device__ __forceinline__ void cvt_store4(_Float16* p, f4 v) {
    h4 h;
    h[0] = (_Float16)v[0]; h[1] = (_Float16)v[1];
    h[2] = (_Float16)v[2]; h[3] = (_Float16)v[3];
    *reinterpret_cast<h4*>(p) = h;
}

// S: 16 x (KT*32) f16 row-major A tile in LDS.  Bp: pre-packed B fragments,
// layout: [kt][ntile j][lane][8 dwords], lane n = j*16 + (lane&15),
// k = kt*32 + (lane>>4)*16 + 2v{,+1} packed per dword (contiguous k order, same as the A reads).
template <int KT>
__device__ __forceinline__ void wmma_tile_gemm(const _Float16* S,
                                               const uint32_t* Bp,
                                               v8f acc[8], int lane) {
    const int m  = lane & 15;
    const int hh = lane >> 4;
    const _Float16* arow = S + m * (KT * 32) + hh * 16;
#pragma unroll
    for (int kt = 0; kt < KT; ++kt) {
        FragH a;
        a.q[0] = *reinterpret_cast<const u32x4*>(arow + kt * 32);
        a.q[1] = *reinterpret_cast<const u32x4*>(arow + kt * 32 + 8);
        const uint32_t* bp = Bp + (size_t)(kt * 256 + lane) * 8;
#pragma unroll
        for (int j = 0; j < 8; ++j) {
            FragH b;
            b.q[0] = *reinterpret_cast<const u32x4*>(bp);
            b.q[1] = *reinterpret_cast<const u32x4*>(bp + 4);
            bp += 256;
            acc[j] = __builtin_amdgcn_wmma_f32_16x16x32_f16(
                false, a.v, false, b.v, (short)0, acc[j], false, false);
        }
    }
}

// Re-pack a f32 row-major [kdim x 128] weight matrix into f16 WMMA B-fragment
// order so each lane's B tile is 32 contiguous bytes.
__global__ void gnn_prepack_kernel(const float* __restrict__ W,
                                   uint32_t* __restrict__ P, int kdim) {
    const int d = blockIdx.x * blockDim.x + threadIdx.x;
    if (d >= kdim * 64) return;
    const int v  = d & 7;
    const int l  = (d >> 3) & 31;
    const int j  = (d >> 8) & 7;
    const int kt = d >> 11;
    const int k0 = kt * 32 + ((l >> 4) << 4) + 2 * v;
    const int n  = j * 16 + (l & 15);
    union { _Float16 h[2]; uint32_t u; } pk;
    pk.h[0] = (_Float16)W[(size_t)k0 * H + n];
    pk.h[1] = (_Float16)W[(size_t)(k0 + 1) * H + n];
    P[d] = pk.u;
}

__global__ void gnn_count_kernel(const int* __restrict__ dst,
                                 float* __restrict__ cnt) {
    const int e = blockIdx.x * blockDim.x + threadIdx.x;
    if (e < N_EDGES) atomicAdd(cnt + dst[e], 1.0f);
}

// Fused edge update: gather + concat-MLP(3H->H->H, ELU) + residual + LN +
// scatter-add of the NEW edge features into the aggregation numerator.
// One wave32 per 16 edges; 4 waves/block.
__global__ __launch_bounds__(128) void gnn_edge_kernel(
    const float* x, const int* __restrict__ src, const int* __restrict__ dst,
    const float* ea_in, float* ea_out,
    const uint32_t* __restrict__ w0p, const uint32_t* __restrict__ w1p,
    const float* __restrict__ b0, const float* __restrict__ b1,
    const float* __restrict__ lng, const float* __restrict__ lnb,
    float* __restrict__ agg) {
    __shared__ __align__(16) _Float16 lds[4][16 * 384];
    const int wave = threadIdx.x >> 5;
    const int lane = threadIdx.x & 31;
    const int e0   = (blockIdx.x * 4 + wave) * 16;
    _Float16* S = lds[wave];

    // ---- stage concat([x[src], x[dst], edge_attr]) as f16: 16 x 384
    for (int r = 0; r < 16; ++r) {
        const int e = e0 + r;
        const int s = src[e];
        const int d = dst[e];
        f4 vs = *reinterpret_cast<const f4*>(x + (size_t)s * H + lane * 4);
        f4 vd = *reinterpret_cast<const f4*>(x + (size_t)d * H + lane * 4);
        f4 ve = *reinterpret_cast<const f4*>(ea_in + (size_t)e * H + lane * 4);
        cvt_store4(S + r * 384 +       lane * 4, vs);
        cvt_store4(S + r * 384 + 128 + lane * 4, vd);
        cvt_store4(S + r * 384 + 256 + lane * 4, ve);
    }
    const int l16 = lane & 15;
    const int lh  = lane >> 4;

    // ---- GEMM1: [16x384] @ [384x128] + b0
    v8f acc[8];
#pragma unroll
    for (int j = 0; j < 8; ++j) {
        const float bv = b0[j * 16 + l16];
#pragma unroll
        for (int r = 0; r < 8; ++r) acc[j][r] = bv;
    }
    wmma_tile_gemm<12>(S, w0p, acc, lane);

    // ---- ELU, write hidden tile (16x128 f16) into LDS (overlays stage)
#pragma unroll
    for (int j = 0; j < 8; ++j)
#pragma unroll
        for (int r = 0; r < 8; ++r) {
            float v = acc[j][r];
            v = v > 0.f ? v : expm1f(v);
            S[(8 * lh + r) * 128 + j * 16 + l16] = (_Float16)v;
        }

    // ---- GEMM2: [16x128] @ [128x128] + b1
    v8f acc2[8];
#pragma unroll
    for (int j = 0; j < 8; ++j) {
        const float bv = b1[j * 16 + l16];
#pragma unroll
        for (int r = 0; r < 8; ++r) acc2[j][r] = bv;
    }
    wmma_tile_gemm<4>(S, w1p, acc2, lane);

    // ---- residual + LayerNorm + store + scatter into aggregation
    float g[8], bb[8];
#pragma unroll
    for (int j = 0; j < 8; ++j) {
        g[j]  = lng[j * 16 + l16];
        bb[j] = lnb[j * 16 + l16];
    }
#pragma unroll
    for (int r = 0; r < 8; ++r) {
        const int m = 8 * lh + r;
        const int e = e0 + m;
#pragma unroll
        for (int j = 0; j < 8; ++j)
            acc2[j][r] += ea_in[(size_t)e * H + j * 16 + l16];
        float s = 0.f, sq = 0.f;
#pragma unroll
        for (int j = 0; j < 8; ++j) {
            const float v = acc2[j][r];
            s += v; sq += v * v;
        }
#pragma unroll
        for (int mask = 1; mask < 16; mask <<= 1) {
            s  += __shfl_xor(s,  mask, 32);
            sq += __shfl_xor(sq, mask, 32);
        }
        const float mean = s * (1.f / 128.f);
        const float var  = sq * (1.f / 128.f) - mean * mean;
        const float rstd = rsqrtf(var + 1e-5f);
        const int dm = dst[e];
#pragma unroll
        for (int j = 0; j < 8; ++j) {
            const int n = j * 16 + l16;
            const float o = (acc2[j][r] - mean) * rstd * g[j] + bb[j];
            ea_out[(size_t)e * H + n] = o;
            atomicAdd(agg + (size_t)dm * H + n, o);
        }
    }
}

// Fused node update: concat(x, agg*inv_cnt) MLP(2H->H->H, ELU) + residual + LN
__global__ __launch_bounds__(128) void gnn_node_kernel(
    const float* x, const float* __restrict__ agg, const float* __restrict__ cnt,
    const uint32_t* __restrict__ w0p, const uint32_t* __restrict__ w1p,
    const float* __restrict__ b0, const float* __restrict__ b1,
    const float* __restrict__ lng, const float* __restrict__ lnb,
    float* x_out) {
    __shared__ __align__(16) _Float16 lds[4][16 * 256];
    const int wave = threadIdx.x >> 5;
    const int lane = threadIdx.x & 31;
    const int n0   = (blockIdx.x * 4 + wave) * 16;
    _Float16* S = lds[wave];

    for (int r = 0; r < 16; ++r) {
        int node = n0 + r;
        if (node >= N_NODES) node = N_NODES - 1;
        const float ic = 1.f / fmaxf(cnt[node], 1.f);
        f4 vx = *reinterpret_cast<const f4*>(x   + (size_t)node * H + lane * 4);
        f4 va = *reinterpret_cast<const f4*>(agg + (size_t)node * H + lane * 4);
        va = va * ic;
        cvt_store4(S + r * 256 +       lane * 4, vx);
        cvt_store4(S + r * 256 + 128 + lane * 4, va);
    }
    const int l16 = lane & 15;
    const int lh  = lane >> 4;

    v8f acc[8];
#pragma unroll
    for (int j = 0; j < 8; ++j) {
        const float bv = b0[j * 16 + l16];
#pragma unroll
        for (int r = 0; r < 8; ++r) acc[j][r] = bv;
    }
    wmma_tile_gemm<8>(S, w0p, acc, lane);

#pragma unroll
    for (int j = 0; j < 8; ++j)
#pragma unroll
        for (int r = 0; r < 8; ++r) {
            float v = acc[j][r];
            v = v > 0.f ? v : expm1f(v);
            S[(8 * lh + r) * 128 + j * 16 + l16] = (_Float16)v;
        }

    v8f acc2[8];
#pragma unroll
    for (int j = 0; j < 8; ++j) {
        const float bv = b1[j * 16 + l16];
#pragma unroll
        for (int r = 0; r < 8; ++r) acc2[j][r] = bv;
    }
    wmma_tile_gemm<4>(S, w1p, acc2, lane);

    float g[8], bb[8];
#pragma unroll
    for (int j = 0; j < 8; ++j) {
        g[j]  = lng[j * 16 + l16];
        bb[j] = lnb[j * 16 + l16];
    }
#pragma unroll
    for (int r = 0; r < 8; ++r) {
        const int node = n0 + 8 * lh + r;
        const int nc   = node < N_NODES ? node : (N_NODES - 1);
#pragma unroll
        for (int j = 0; j < 8; ++j)
            acc2[j][r] += x[(size_t)nc * H + j * 16 + l16];
        float s = 0.f, sq = 0.f;
#pragma unroll
        for (int j = 0; j < 8; ++j) {
            const float v = acc2[j][r];
            s += v; sq += v * v;
        }
#pragma unroll
        for (int mask = 1; mask < 16; mask <<= 1) {
            s  += __shfl_xor(s,  mask, 32);
            sq += __shfl_xor(sq, mask, 32);
        }
        const float mean = s * (1.f / 128.f);
        const float var  = sq * (1.f / 128.f) - mean * mean;
        const float rstd = rsqrtf(var + 1e-5f);
        const bool valid = node < N_NODES;
#pragma unroll
        for (int j = 0; j < 8; ++j) {
            const int n = j * 16 + l16;
            const float o = (acc2[j][r] - mean) * rstd * g[j] + bb[j];
            if (valid) x_out[(size_t)node * H + n] = o;
        }
    }
}

__global__ void gnn_out_kernel(const float* __restrict__ x,
                               const float* __restrict__ w,
                               const float* __restrict__ b,
                               float* __restrict__ out) {
    const int n = blockIdx.x * blockDim.x + threadIdx.x;
    if (n >= N_NODES) return;
    float a0 = b[0], a1 = b[1], a2 = b[2];
    const float* xr = x + (size_t)n * H;
#pragma unroll 8
    for (int h = 0; h < H; ++h) {
        const float v = xr[h];
        a0 += v * w[h * 3 + 0];
        a1 += v * w[h * 3 + 1];
        a2 += v * w[h * 3 + 2];
    }
    a0 = a0 > 0.f ? a0 : expm1f(a0);
    a1 = a1 > 0.f ? a1 : expm1f(a1);
    a2 = a2 > 0.f ? a2 : expm1f(a2);
    out[(size_t)n * 3 + 0] = a0;
    out[(size_t)n * 3 + 1] = a1;
    out[(size_t)n * 3 + 2] = a2;
}

extern "C" void kernel_launch(void* const* d_in, const int* in_sizes, int n_in,
                              void* d_out, int out_size, void* d_ws, size_t ws_size,
                              hipStream_t stream) {
    const float* x_in  = (const float*)d_in[0];
    const int*   eidx  = (const int*)d_in[1];
    const float* ea0   = (const float*)d_in[2];
    const float* ew0   = (const float*)d_in[3];
    const float* eb0   = (const float*)d_in[4];
    const float* ew1   = (const float*)d_in[5];
    const float* eb1   = (const float*)d_in[6];
    const float* elg   = (const float*)d_in[7];
    const float* elb   = (const float*)d_in[8];
    const float* nw0   = (const float*)d_in[9];
    const float* nb0   = (const float*)d_in[10];
    const float* nw1   = (const float*)d_in[11];
    const float* nb1   = (const float*)d_in[12];
    const float* nlg   = (const float*)d_in[13];
    const float* nlb   = (const float*)d_in[14];
    const float* outw  = (const float*)d_in[15];
    const float* outb  = (const float*)d_in[16];
    const int* src = eidx;
    const int* dst = eidx + N_EDGES;

    char* ws = (char*)d_ws;
    float*    xb   = (float*)(ws);                          // N*H*4   = 25,600,000
    float*    eab  = (float*)(ws + 25600000ull);            // E*H*4   = 409,600,000
    float*    agg  = (float*)(ws + 435200000ull);           // N*H*4   = 25,600,000
    float*    cnt  = (float*)(ws + 460800000ull);           // N*4     (padded)
    uint32_t* pk   = (uint32_t*)(ws + 461000192ull);        // packed weights
    uint32_t* ew0p = pk;                                    // 2 * 24576 dwords
    uint32_t* ew1p = ew0p + 2 * 24576;                      // 2 *  8192
    uint32_t* nw0p = ew1p + 2 * 8192;                       // 2 * 16384
    uint32_t* nw1p = nw0p + 2 * 16384;                      // 2 *  8192

    // Pre-pack all weight matrices into WMMA B-fragment order (tiny; L2-hot).
    for (int i = 0; i < NLAYERS; ++i) {
        gnn_prepack_kernel<<<96, 256, 0, stream>>>(ew0 + (size_t)i * 384 * H, ew0p + i * 24576, 384);
        gnn_prepack_kernel<<<32, 256, 0, stream>>>(ew1 + (size_t)i * 128 * H, ew1p + i * 8192, 128);
        gnn_prepack_kernel<<<64, 256, 0, stream>>>(nw0 + (size_t)i * 256 * H, nw0p + i * 16384, 256);
        gnn_prepack_kernel<<<32, 256, 0, stream>>>(nw1 + (size_t)i * 128 * H, nw1p + i * 8192, 128);
    }

    hipMemsetAsync(cnt, 0, N_NODES * sizeof(float), stream);
    gnn_count_kernel<<<(N_EDGES + 255) / 256, 256, 0, stream>>>(dst, cnt);

    for (int i = 0; i < NLAYERS; ++i) {
        hipMemsetAsync(agg, 0, (size_t)N_NODES * H * sizeof(float), stream);
        const float* xcur = (i == 0) ? x_in : xb;
        const float* eac  = (i == 0) ? ea0 : eab;   // in-place after layer 0 (row-owned)
        gnn_edge_kernel<<<N_EDGES / 64, 128, 0, stream>>>(
            xcur, src, dst, eac, eab, ew0p + i * 24576, ew1p + i * 8192,
            eb0 + i * H, eb1 + i * H, elg + i * H, elb + i * H, agg);
        gnn_node_kernel<<<(N_NODES + 63) / 64, 128, 0, stream>>>(
            xcur, agg, cnt, nw0p + i * 16384, nw1p + i * 8192,
            nb0 + i * H, nb1 + i * H, nlg + i * H, nlb + i * H, xb);
    }

    gnn_out_kernel<<<(N_NODES + 255) / 256, 256, 0, stream>>>(xb, outw, outb, (float*)d_out);
}
